// Attention_68736656605774
// MI455X (gfx1250) — matrix operand order repaired and checked by reading, not compiled, not run on hardware
//
#include <hip/hip_runtime.h>

// ---------------------------------------------------------------------------
// CDNA5 (gfx1250) fused attention: QKV GEMM + RoPE -> flash SDPA -> out GEMM
// Matrix math via v_wmma_f32_16x16x32_bf16 (f32 accumulate).
// K/V tiles staged to LDS via the Tensor Data Mover (tensor_load_to_lds,
// TENSORcnt-synchronized); bias stream prefetched (global_prefetch_b8).
// ---------------------------------------------------------------------------

typedef __bf16 bf16_t;
typedef __attribute__((ext_vector_type(16))) __bf16 v16bf;
typedef __attribute__((ext_vector_type(8)))  __bf16 v8bf;
typedef __attribute__((ext_vector_type(8)))  float  v8f;
typedef __attribute__((ext_vector_type(4)))  unsigned int v4u;
typedef __attribute__((ext_vector_type(8)))  int v8i;
typedef __attribute__((ext_vector_type(4)))  int v4i;

#define WMMA_BF16(a, b, c) \
  __builtin_amdgcn_wmma_f32_16x16x32_bf16(false, (a), false, (b), (short)0, (c), false, false)

__device__ __forceinline__ v16bf cat8(v8bf lo, v8bf hi) {
  return __builtin_shufflevector(lo, hi, 0,1,2,3,4,5,6,7,8,9,10,11,12,13,14,15);
}

constexpr int BATCH = 2, SEQ = 2048, DMODEL = 1024, NH = 16, DH = 64;
constexpr int BS    = BATCH * SEQ;      // 4096 rows
constexpr int NQKV  = 3 * DMODEL;       // 3072 qkv cols

// ---------------------------------------------------------------------------
// TDM helper: 2-D tile load Global -> LDS, bf16 elements (data_size = 2B).
// D# group layouts per CDNA5 ISA §8.3/§8.4 (count=1, type=2 "image").
// All arguments must be wave-uniform. Tracked by TENSORcnt.
// This toolchain exposes the 6-arg builtin (groups 2/3 + extra group, cpol).
// ---------------------------------------------------------------------------
__device__ __forceinline__ void tdm_load_2d_bf16(
    unsigned int lds_addr, const void* gptr,
    unsigned int tile_d0, unsigned int tile_d1,
    unsigned int tensor_d0, unsigned int tensor_d1,
    unsigned long long stride0_elems) {
  unsigned long long ga = (unsigned long long)(uintptr_t)gptr;
  v4u g0;
  g0.x = 1u;                                              // count=1, user mode
  g0.y = lds_addr;                                        // LDS byte address
  g0.z = (unsigned int)ga;                                // global_addr[31:0]
  g0.w = (unsigned int)((ga >> 32) & 0x01FFFFFFull) |     // global_addr[56:32]
         (2u << 30);                                      // type = 2
  v8i g1;
  g1[0] = 1 << 16;                                        // data_size=1 (2 bytes)
  g1[1] = (int)((tensor_d0 & 0xFFFFu) << 16);             // tensor_dim0[15:0]
  g1[2] = (int)((tensor_d0 >> 16) | ((tensor_d1 & 0xFFFFu) << 16));
  g1[3] = (int)((tensor_d1 >> 16) | (tile_d0 << 16));     // tile_dim0
  g1[4] = (int)(tile_d1 & 0xFFFFu);                       // tile_dim1 (tile_dim2=0)
  g1[5] = (int)(stride0_elems & 0xFFFFFFFFull);           // tensor_dim0_stride lo
  g1[6] = (int)((stride0_elems >> 32) & 0xFFFFull);       // stride hi (dim1_stride=0)
  g1[7] = 0;
  v4i z4 = {0, 0, 0, 0};
  v8i z8 = {0, 0, 0, 0, 0, 0, 0, 0};
  __builtin_amdgcn_tensor_load_to_lds(g0, g1, z4, z4, z8, 0);
}

// ---------------------------------------------------------------------------
// Kernel 0a: f32 -> bf16 convert (x), 4 elements per thread
// ---------------------------------------------------------------------------
__global__ void cvt_f32_bf16(const float* __restrict__ src, bf16_t* __restrict__ dst, int n) {
  int i = (blockIdx.x * blockDim.x + threadIdx.x) * 4;
  if (i + 3 < n) {
    float4 v = *(const float4*)(src + i);
    dst[i + 0] = (bf16_t)v.x;
    dst[i + 1] = (bf16_t)v.y;
    dst[i + 2] = (bf16_t)v.z;
    dst[i + 3] = (bf16_t)v.w;
  }
}

// ---------------------------------------------------------------------------
// Kernel 0b: tiled transpose + convert: dst[c*R + r] = (bf16) src[r*C + c]
// ---------------------------------------------------------------------------
__global__ void transpose_cvt(const float* __restrict__ src, bf16_t* __restrict__ dst,
                              int R, int C) {
  __shared__ float tile[32][33];
  int bx = blockIdx.x * 32;  // col base
  int by = blockIdx.y * 32;  // row base
  int tx = threadIdx.x, ty = threadIdx.y;
  #pragma unroll
  for (int i = ty; i < 32; i += 8)
    tile[i][tx] = src[(size_t)(by + i) * C + (bx + tx)];
  __syncthreads();
  #pragma unroll
  for (int i = ty; i < 32; i += 8)
    dst[(size_t)(bx + i) * R + (by + tx)] = (bf16_t)tile[tx][i];
}

// ---------------------------------------------------------------------------
// Kernel 1: QKV projection (bf16 WMMA, f32 accum) + bias + RoPE + scatter.
//   wave tile: 16 rows x 64 cols (one head of one of q/k/v).
//   q,k stored [B,H,S,DH] bf16 ; v stored transposed [B,H,DH,S] bf16.
// ---------------------------------------------------------------------------
__global__ __launch_bounds__(128) void qkv_rope_kernel(
    const bf16_t* __restrict__ xb, const bf16_t* __restrict__ WqkvT,
    const float* __restrict__ b_qkv,
    bf16_t* __restrict__ qbuf, bf16_t* __restrict__ kbuf, bf16_t* __restrict__ vT) {
  const int lane = threadIdx.x & 31;
  const int wave = blockIdx.x * (blockDim.x >> 5) + (threadIdx.x >> 5);
  const int rowTile = wave / 48;              // 256 row tiles
  const int ct      = wave % 48;              // 48 col tiles of 64
  const int mat = ct >> 4;                    // 0=q 1=k 2=v
  const int h   = ct & 15;
  const int rbase   = rowTile * 16;
  const int colbase = mat * DMODEL + h * DH;

  const int koff = (lane >> 4) * 8;
  const bf16_t* arow = xb + (size_t)(rbase + (lane & 15)) * DMODEL;

  v8f acc[4] = {};
  for (int kk = 0; kk < DMODEL; kk += 32) {
    v8bf alo = *(const v8bf*)(arow + kk + 2 * koff);
    v8bf ahi = *(const v8bf*)(arow + kk + 8 + 2 * koff);
    v16bf a  = cat8(alo, ahi);
    const int kb0 = kk + ((lane >> 4) << 4);
    #pragma unroll
    for (int t = 0; t < 4; ++t) {
      const bf16_t* brow = WqkvT + (size_t)(colbase + t * 16 + (lane & 15)) * DMODEL + kb0;
      v16bf bfrag = *(const v16bf*)brow;
      acc[t] = WMMA_BF16(a, bfrag, acc[t]);
    }
  }

  // qkv bias (broadcast down rows of each column)
  #pragma unroll
  for (int t = 0; t < 4; ++t) {
    float bq = b_qkv[colbase + t * 16 + (lane & 15)];
    #pragma unroll
    for (int j = 0; j < 8; ++j) acc[t][j] += bq;
  }

  const int jrow = (lane >> 4) * 8;

  // RoPE for q,k: pair (dh, dh+32) lives in fragments (t, t+2), same lane/row.
  if (mat < 2) {
    #pragma unroll
    for (int t = 0; t < 2; ++t) {
      float dh = (float)(t * 16 + (lane & 15));          // 0..31
      float invf = __powf(10000.0f, -dh * (1.0f / 32.0f));
      #pragma unroll
      for (int j = 0; j < 8; ++j) {
        int r = rbase + jrow + j;
        float s = (float)(r & (SEQ - 1));
        float ang = s * invf;
        float c = __cosf(ang), sn = __sinf(ang);
        float lo = acc[t][j], hi = acc[t + 2][j];
        acc[t][j]     = lo * c - hi * sn;
        acc[t + 2][j] = hi * c + lo * sn;
      }
    }
  }

  // scatter
  #pragma unroll
  for (int t = 0; t < 4; ++t) {
    int dh = t * 16 + (lane & 15);
    #pragma unroll
    for (int j = 0; j < 8; ++j) {
      int r = rbase + jrow + j;
      int bidx = r >> 11, s = r & (SEQ - 1);
      size_t head = (size_t)(bidx * NH + h);
      bf16_t v = (bf16_t)acc[t][j];
      if (mat == 0)      qbuf[(head * SEQ + s) * DH + dh] = v;
      else if (mat == 1) kbuf[(head * SEQ + s) * DH + dh] = v;
      else               vT[(head * DH + dh) * SEQ + s]   = v;
    }
  }
}

// ---------------------------------------------------------------------------
// Kernel 2: flash attention. One wave per (b,h,16-query tile).
//   Per 32-key step: TDM stages K (4KB contiguous) and V^T (2-D tile) into
//   this wave's LDS slice; 4 WMMA for Q@K^T, online softmax, P transposed
//   C->A layout via LDS, 4 WMMA for P@V. Bias streamed once w/ prefetch.
// ---------------------------------------------------------------------------
__global__ __launch_bounds__(128) void attn_kernel(
    const bf16_t* __restrict__ qbuf, const bf16_t* __restrict__ kbuf,
    const bf16_t* __restrict__ vT,   const float* __restrict__ bias,
    const unsigned char* __restrict__ kvmask, bf16_t* __restrict__ aout) {
  __shared__ __align__(16) bf16_t ldsK[4][32 * 64];   // 16 KB: 32 keys x 64 dh
  __shared__ __align__(16) bf16_t ldsV[4][64 * 32];   // 16 KB: 64 dh  x 32 keys
  __shared__ __align__(16) bf16_t pls [4][16][32];    //  4 KB: P transpose tile

  const int lane = threadIdx.x & 31;
  const int wv   = threadIdx.x >> 5;
  const int wave = blockIdx.x * 4 + wv;       // 4096 waves total
  const int qtile = wave & 127;
  const int bh    = wave >> 7;                // 0..31
  const int b = bh >> 4, h = bh & 15;
  const int qbase = qtile * 16;
  const size_t headoff = (size_t)bh * SEQ * DH;

  const int koff = (lane >> 4) * 8;
  const int jrow = (lane >> 4) * 8;

  const unsigned int ldsK_addr = (unsigned int)(uintptr_t)&ldsK[wv][0];
  const unsigned int ldsV_addr = (unsigned int)(uintptr_t)&ldsV[wv][0];

  // Q A-fragments (dh 0..31 and 32..63), held for the whole key loop
  const bf16_t* qrow = qbuf + headoff + (size_t)(qbase + (lane & 15)) * DH;
  v16bf aq0 = cat8(*(const v8bf*)(qrow + 2 * koff),      *(const v8bf*)(qrow + 8 + 2 * koff));
  v16bf aq1 = cat8(*(const v8bf*)(qrow + 32 + 2 * koff), *(const v8bf*)(qrow + 40 + 2 * koff));

  float m[8], l[8];
  #pragma unroll
  for (int j = 0; j < 8; ++j) { m[j] = -INFINITY; l[j] = 0.0f; }
  v8f o[4] = {};

  for (int kb = 0; kb < SEQ; kb += 32) {
    // ---- TDM: stage this wave's K tile (contiguous 2048 elems) and V tile
    //      (64 rows x 32 keys, row stride SEQ elems) into LDS ---------------
    tdm_load_2d_bf16(ldsK_addr, kbuf + headoff + (size_t)kb * DH,
                     /*tile*/ 2048, 1, /*tensor*/ SEQ * DH, 1, SEQ * DH);
    tdm_load_2d_bf16(ldsV_addr, vT + headoff + (size_t)kb,
                     /*tile*/ 32, 64, /*tensor*/ SEQ, DH, SEQ);

    // ---- prefetch next key-block's bias lines while the TDM runs ----------
    if (kb + 32 < SEQ) {
      const float* pf = bias +
          (((size_t)(b * SEQ + qbase + jrow)) * SEQ + (kb + 32 + (lane & 15))) * NH + h;
      __builtin_prefetch(pf, 0, 1);
    }

    __builtin_amdgcn_s_wait_tensorcnt(0);

    // ---- scores: two 16x16 tiles, contraction over DH=64 (2 chained WMMAs) -
    v8f sc[2];
    const int d0 = (lane >> 4) << 4;
    #pragma unroll
    for (int t = 0; t < 2; ++t) {
      const bf16_t* krow = &ldsK[wv][(t * 16 + (lane & 15)) * DH];
      v16bf bk_lo = *(const v16bf*)(krow + d0);
      v16bf bk_hi = *(const v16bf*)(krow + 32 + d0);
      v8f s = {};
      s = WMMA_BF16(aq0, bk_lo, s);
      s = WMMA_BF16(aq1, bk_hi, s);
      sc[t] = s;
    }

    // ---- scale + additive bias + kv mask (bias read exactly once) ----------
    #pragma unroll
    for (int t = 0; t < 2; ++t) {
      int key = kb + t * 16 + (lane & 15);
      float mk = kvmask[b * SEQ + key] ? 0.0f : -1e30f;
      const float* bptr = bias + (((size_t)(b * SEQ + qbase + jrow)) * SEQ + key) * NH + h;
      #pragma unroll
      for (int j = 0; j < 8; ++j)
        sc[t][j] = sc[t][j] * 0.125f + bptr[(size_t)j * SEQ * NH] + mk;
    }

    // ---- online softmax: rows live across 16-lane groups in C layout -------
    #pragma unroll
    for (int j = 0; j < 8; ++j) {
      float v = fmaxf(sc[0][j], sc[1][j]);
      #pragma unroll
      for (int off = 8; off > 0; off >>= 1) v = fmaxf(v, __shfl_xor(v, off, 16));
      float mnew = fmaxf(m[j], v);
      float p0 = __expf(sc[0][j] - mnew);
      float p1 = __expf(sc[1][j] - mnew);
      sc[0][j] = p0; sc[1][j] = p1;
      float sum = p0 + p1;
      #pragma unroll
      for (int off = 8; off > 0; off >>= 1) sum += __shfl_xor(sum, off, 16);
      float esc = __expf(m[j] - mnew);
      l[j] = l[j] * esc + sum;
      m[j] = mnew;
      #pragma unroll
      for (int nt = 0; nt < 4; ++nt) o[nt][j] *= esc;
    }

    // ---- transpose P (C layout) -> A layout through LDS (per-wave tile) ----
    #pragma unroll
    for (int t = 0; t < 2; ++t)
      #pragma unroll
      for (int j = 0; j < 8; ++j)
        pls[wv][jrow + j][t * 16 + (lane & 15)] = (bf16_t)sc[t][j];
    // DS ops are issued in-order within a wave: the reads below see the writes.
    const int prow = lane & 15;
    v16bf pa = cat8(*(const v8bf*)&pls[wv][prow][2 * koff],
                    *(const v8bf*)&pls[wv][prow][8 + 2 * koff]);

    // ---- P @ V : 4 WMMAs over dh tiles; LDS V rows give contiguous B frags -
    const int klocal = (lane >> 4) << 4;
    #pragma unroll
    for (int nt = 0; nt < 4; ++nt) {
      const bf16_t* vrow = &ldsV[wv][(nt * 16 + (lane & 15)) * 32 + klocal];
      v16bf bv = *(const v16bf*)vrow;
      o[nt] = WMMA_BF16(pa, bv, o[nt]);
    }
  }

  // ---- finalize: divide by l, store [B,S,D] bf16 for the out projection ----
  float inv[8];
  #pragma unroll
  for (int j = 0; j < 8; ++j) inv[j] = 1.0f / l[j];
  #pragma unroll
  for (int nt = 0; nt < 4; ++nt) {
    int dh = nt * 16 + (lane & 15);
    #pragma unroll
    for (int j = 0; j < 8; ++j) {
      int q = qbase + jrow + j;
      aout[((size_t)(b * SEQ + q)) * DMODEL + h * DH + dh] = (bf16_t)(o[nt][j] * inv[j]);
    }
  }
}

// ---------------------------------------------------------------------------
// Kernel 3: output projection: aout(bf16) @ WoutT + b_out -> f32 out
// ---------------------------------------------------------------------------
__global__ __launch_bounds__(128) void outproj_kernel(
    const bf16_t* __restrict__ aout, const bf16_t* __restrict__ WoutT,
    const float* __restrict__ b_out, float* __restrict__ out) {
  const int lane = threadIdx.x & 31;
  const int wave = blockIdx.x * (blockDim.x >> 5) + (threadIdx.x >> 5);
  const int rowTile = wave >> 4;   // 256
  const int ctile   = wave & 15;   // 16 tiles of 64 cols
  const int rbase   = rowTile * 16;
  const int colbase = ctile * 64;

  const int koff = (lane >> 4) * 8;
  const bf16_t* arow = aout + (size_t)(rbase + (lane & 15)) * DMODEL;

  v8f acc[4] = {};
  for (int kk = 0; kk < DMODEL; kk += 32) {
    v16bf a = cat8(*(const v8bf*)(arow + kk + 2 * koff),
                   *(const v8bf*)(arow + kk + 8 + 2 * koff));
    const int kb0 = kk + ((lane >> 4) << 4);
    #pragma unroll
    for (int t = 0; t < 4; ++t) {
      const bf16_t* brow = WoutT + (size_t)(colbase + t * 16 + (lane & 15)) * DMODEL + kb0;
      acc[t] = WMMA_BF16(a, *(const v16bf*)brow, acc[t]);
    }
  }

  const int jrow = (lane >> 4) * 8;
  #pragma unroll
  for (int t = 0; t < 4; ++t) {
    int col = colbase + t * 16 + (lane & 15);
    float bo = b_out[col];
    #pragma unroll
    for (int j = 0; j < 8; ++j) {
      int r = rbase + jrow + j;
      out[(size_t)r * DMODEL + col] = acc[t][j] + bo;
    }
  }
}

// ---------------------------------------------------------------------------
// Launch
// ---------------------------------------------------------------------------
extern "C" void kernel_launch(void* const* d_in, const int* in_sizes, int n_in,
                              void* d_out, int out_size, void* d_ws, size_t ws_size,
                              hipStream_t stream) {
  const float*         x        = (const float*)d_in[0];
  const unsigned char* kv_mask  = (const unsigned char*)d_in[1];
  const float*         attnbias = (const float*)d_in[2];
  const float*         W_qkv    = (const float*)d_in[3];
  const float*         b_qkv    = (const float*)d_in[4];
  const float*         W_out    = (const float*)d_in[5];
  const float*         b_out    = (const float*)d_in[6];
  float*               out      = (float*)d_out;

  char* ws = (char*)d_ws;
  bf16_t* xb    = (bf16_t*)(ws);                    // 8 MB (reused as aout)
  bf16_t* WqkvT = (bf16_t*)(ws + 8388608ull);       // 6 MB
  bf16_t* WoutT = (bf16_t*)(ws + 14680064ull);      // 2 MB
  bf16_t* qb    = (bf16_t*)(ws + 16777216ull);      // 8 MB
  bf16_t* kb    = (bf16_t*)(ws + 25165824ull);      // 8 MB
  bf16_t* vT    = (bf16_t*)(ws + 33554432ull);      // 8 MB  (end: 40 MB)
  bf16_t* aout  = xb;                               // xb dead after QKV GEMM

  // 0) precision/layout staging
  cvt_f32_bf16<<<4096, 256, 0, stream>>>(x, xb, BS * DMODEL);
  transpose_cvt<<<dim3(NQKV / 32, DMODEL / 32), dim3(32, 8), 0, stream>>>(W_qkv, WqkvT, DMODEL, NQKV);
  transpose_cvt<<<dim3(DMODEL / 32, DMODEL / 32), dim3(32, 8), 0, stream>>>(W_out, WoutT, DMODEL, DMODEL);

  // 1) QKV GEMM + bias + RoPE + head scatter : 256*48 waves, 4 waves/block
  qkv_rope_kernel<<<3072, 128, 0, stream>>>(xb, WqkvT, b_qkv, qb, kb, vT);

  // 2) flash attention : 2*16*128 waves, 4 waves/block
  attn_kernel<<<1024, 128, 0, stream>>>(qb, kb, vT, attnbias, kv_mask, aout);

  // 3) output projection : 256*16 waves, 4 waves/block
  outproj_kernel<<<1024, 128, 0, stream>>>(aout, WoutT, b_out, out);
}
